// GNNEncoder_20289425506400
// MI455X (gfx1250) — matrix operand order repaired and checked by reading, not compiled, not run on hardware
//
#include <hip/hip_runtime.h>
#include <hip/hip_bf16.h>

typedef float v8f __attribute__((ext_vector_type(8)));
typedef __bf16 bf16t;
typedef bf16t v16bf __attribute__((ext_vector_type(16)));

#define GD_IN  128
#define GD_H   128
#define GD_OUT 64
#define GBN_EPS 1e-5f

// ---------------- bf16 split helpers (manual RNE, bit ops only) ----------------
__device__ inline unsigned short bf16_rne(float f) {
    unsigned u = __float_as_uint(f);
    unsigned r = u + 0x7fffu + ((u >> 16) & 1u);
    return (unsigned short)(r >> 16);
}
__device__ inline float bf16_to_f(unsigned short h) {
    return __uint_as_float(((unsigned)h) << 16);
}

// ---------------- utility: fill ----------------
__global__ void fill_kernel(float* __restrict__ p, float v, size_t n) {
    size_t i = (size_t)blockIdx.x * blockDim.x + threadIdx.x;
    size_t stride = (size_t)gridDim.x * blockDim.x;
    for (; i < n; i += stride) p[i] = v;
}

// ---------------- degree / dinv ----------------
__global__ void deg_kernel(const int* __restrict__ dst, const float* __restrict__ w,
                           float* __restrict__ deg, int E) {
    int e = blockIdx.x * blockDim.x + threadIdx.x;
    if (e < E) atomicAdd(&deg[dst[e]], w[e]);
}

__global__ void rsqrt_kernel(float* __restrict__ d, int n) {
    int i = blockIdx.x * blockDim.x + threadIdx.x;
    if (i < n) d[i] = 1.0f / sqrtf(d[i]);   // deg >= 1 always (self-loop weight 1)
}

// ---------------- split fp32 -> bf16 hi/lo planes (layer-1 input only) ----------------
__global__ void split_hilo(const float* __restrict__ src, unsigned short* __restrict__ hi,
                           unsigned short* __restrict__ lo, size_t n) {
    size_t i = (size_t)blockIdx.x * blockDim.x + threadIdx.x;
    if (i >= n) return;
    float v = src[i];
    unsigned short h = bf16_rne(v);
    hi[i] = h;
    lo[i] = bf16_rne(v - bf16_to_f(h));
}

// ---------------- pack weights into B-fragment order ----------------
// out[((kb*DCOL + n)*2 + half)*16 + j] = W[(kb*32 + half*16 + j)][n]
// -> each lane's 16 K-values for (column n, lane-half) are contiguous (32B).
template <int DCOL>
__global__ void pack_w(const float* __restrict__ W, unsigned short* __restrict__ hi,
                       unsigned short* __restrict__ lo) {
    int o = blockIdx.x * blockDim.x + threadIdx.x;
    if (o >= GD_IN * DCOL) return;
    int j    = o & 15;
    int half = (o >> 4) & 1;
    int n    = (o >> 5) & (DCOL - 1);
    int kb   = o / (DCOL * 32);
    float v = W[(size_t)(kb * 32 + half * 16 + j) * DCOL + n];
    unsigned short h = bf16_rne(v);
    hi[o] = h;
    lo[o] = bf16_rne(v - bf16_to_f(h));
}

// ---------------- split-precision bf16 WMMA GEMM ----------------
// C[nrows x DCOL] = A @ B, A = Ahi+Alo (row-major bf16 planes, K=128),
// B pre-packed fragment-order planes. acc += Ah*Bh + Ah*Bl + Al*Bh (fp32 acc).
// One wave -> 16-row strip x all DCOL/16 column tiles.
union frag_u { uint4 u[2]; v16bf v; };

template <int DCOL>
__global__ void __launch_bounds__(256) gemm_bf16x3(
    const unsigned short* __restrict__ Ahi, const unsigned short* __restrict__ Alo,
    const unsigned short* __restrict__ Bhi, const unsigned short* __restrict__ Blo,
    float* __restrict__ C, int nrows) {
    const int wave = (int)((blockIdx.x * (size_t)blockDim.x + threadIdx.x) >> 5);
    const int lane = threadIdx.x & 31;
    const int m0 = wave << 4;
    if (m0 >= nrows) return;              // wave-uniform: EXEC all-ones for WMMA
    const int half = lane >> 4;
    const int l = lane & 15;

    constexpr int NT = DCOL / 16;
    v8f acc[NT] = {};

    int arow = m0 + l;
    if (arow >= nrows) arow = nrows - 1;  // clamp (keeps EXEC uniform, avoids OOB reads)
    const uint4* ahr = (const uint4*)(Ahi + (size_t)arow * GD_IN);  // 16 uint4 per row
    const uint4* alr = (const uint4*)(Alo + (size_t)arow * GD_IN);
    const uint4* bhr = (const uint4*)Bhi;
    const uint4* blr = (const uint4*)Blo;

#pragma unroll
    for (int kb = 0; kb < GD_IN / 32; ++kb) {
        const int c0 = kb * 4 + half * 2;       // (kb*32 + half*16)/8
        const int c1 = kb * 4 + half * 2 + 1;   // (kb*32 + half*16 + 8)/8
        frag_u ah, al;
        ah.u[0] = ahr[c0]; ah.u[1] = ahr[c1];
        al.u[0] = alr[c0]; al.u[1] = alr[c1];
#pragma unroll
        for (int t = 0; t < NT; ++t) {
            const size_t base = ((size_t)(kb * DCOL + t * 16 + l) * 2 + half) * 2;  // uint4 units
            frag_u bh, bl;
            bh.u[0] = bhr[base]; bh.u[1] = bhr[base + 1];
            bl.u[0] = blr[base]; bl.u[1] = blr[base + 1];
            acc[t] = __builtin_amdgcn_wmma_f32_16x16x32_bf16(
                false, ah.v, false, bh.v, (short)0, acc[t], false, false);
            acc[t] = __builtin_amdgcn_wmma_f32_16x16x32_bf16(
                false, ah.v, false, bl.v, (short)0, acc[t], false, false);
            acc[t] = __builtin_amdgcn_wmma_f32_16x16x32_bf16(
                false, al.v, false, bh.v, (short)0, acc[t], false, false);
        }
    }
#pragma unroll
    for (int i = 0; i < 8; ++i) {
        const int row = m0 + (half << 3) + i;
        if (row < nrows) {
#pragma unroll
            for (int t = 0; t < NT; ++t)
                C[(size_t)row * DCOL + t * 16 + l] = acc[t][i];
        }
    }
}

// ---------------- edge scatter: out[dst] += h[src] * norm  (one wave per edge) ----------------
template <int D>
__global__ void __launch_bounds__(256) scatter_edges(const int* __restrict__ src,
                                                     const int* __restrict__ dst,
                                                     const float* __restrict__ w,
                                                     const float* __restrict__ dinv,
                                                     const float* __restrict__ h,
                                                     float* __restrict__ out, int E) {
    const int t = blockIdx.x * blockDim.x + threadIdx.x;
    const int edge = t >> 5;
    const int lane = t & 31;
    if (edge >= E) return;
    const int s = src[edge];
    const int d = dst[edge];
    const float norm = dinv[s] * w[edge] * dinv[d];
    constexpr int PER = D / 32;  // 4 (D=128) or 2 (D=64)
    const float* hs = h + (size_t)s * D + lane * PER;
    float* od = out + (size_t)d * D + lane * PER;
    if constexpr (PER == 4) {
        float4 v = *(const float4*)hs;
        atomicAdd(&od[0], v.x * norm);
        atomicAdd(&od[1], v.y * norm);
        atomicAdd(&od[2], v.z * norm);
        atomicAdd(&od[3], v.w * norm);
    } else {
        float2 v = *(const float2*)hs;
        atomicAdd(&od[0], v.x * norm);
        atomicAdd(&od[1], v.y * norm);
    }
}

// ---------------- agg init: y[i,c] = h[i,c]*dinv[i]^2 + b[c]  (self-loop + bias, pre-scatter) ----------------
template <int D>
__global__ void init_agg(float* __restrict__ y, const float* __restrict__ h,
                         const float* __restrict__ dinv, const float* __restrict__ b, int n) {
    size_t i = (size_t)blockIdx.x * blockDim.x + threadIdx.x;
    if (i >= (size_t)n * D) return;
    const int row = (int)(i / D);
    const int c = (int)(i % D);
    const float di = dinv[row];
    y[i] = h[i] * (di * di) + b[c];
}

// ---------------- BatchNorm column stats (sum / sumsq) ----------------
__global__ void bn_stats(const float* __restrict__ x, float* __restrict__ stats, int n) {
    const int c = threadIdx.x;  // 0..127
    const int ROWS = 256;
    int r = blockIdx.x * ROWS + threadIdx.y;
    int rend = blockIdx.x * ROWS + ROWS;
    if (rend > n) rend = n;
    float s = 0.f, s2 = 0.f;
    for (; r < rend; r += 2) {
        float v = x[(size_t)r * 128 + c];
        s += v;
        s2 += v * v;
    }
    atomicAdd(&stats[c], s);
    atomicAdd(&stats[128 + c], s2);
}

// ---------------- fused BatchNorm + ReLU + bf16 hi/lo split ----------------
// fp32 activation never materializes: write next GEMM's A-planes directly.
__global__ void bn_relu_split(const float* __restrict__ x, unsigned short* __restrict__ hi,
                              unsigned short* __restrict__ lo, const float* __restrict__ stats,
                              const float* __restrict__ gamma, const float* __restrict__ beta,
                              int n, float invN) {
    size_t i = (size_t)blockIdx.x * blockDim.x + threadIdx.x;
    if (i >= (size_t)n * 128) return;
    const int c = (int)(i & 127);
    const float mean = stats[c] * invN;
    float var = stats[128 + c] * invN - mean * mean;
    if (var < 0.f) var = 0.f;
    const float scale = gamma[c] * (1.0f / sqrtf(var + GBN_EPS));
    float v = (x[i] - mean) * scale + beta[c];
    v = v > 0.f ? v : 0.f;
    unsigned short h = bf16_rne(v);
    hi[i] = h;
    lo[i] = bf16_rne(v - bf16_to_f(h));
}

extern "C" void kernel_launch(void* const* d_in, const int* in_sizes, int n_in,
                              void* d_out, int out_size, void* d_ws, size_t ws_size,
                              hipStream_t stream) {
    (void)n_in; (void)out_size; (void)ws_size;

    const float* x   = (const float*)d_in[0];
    const int*   ei  = (const int*)d_in[1];
    const float* ew  = (const float*)d_in[2];
    const float* W1  = (const float*)d_in[3];
    const float* b1  = (const float*)d_in[4];
    const float* W2  = (const float*)d_in[5];
    const float* b2  = (const float*)d_in[6];
    const float* W3  = (const float*)d_in[7];
    const float* b3  = (const float*)d_in[8];
    const float* g1  = (const float*)d_in[9];
    const float* be1 = (const float*)d_in[10];
    const float* g2  = (const float*)d_in[11];
    const float* be2 = (const float*)d_in[12];

    const int N = in_sizes[0] / GD_IN;
    const int E = in_sizes[1] / 2;
    const int* src = ei;
    const int* dst = ei + E;

    float* ws    = (float*)d_ws;
    float* dinv  = ws;                           // N floats
    float* bufA  = dinv + N;                     // N*128 floats
    float* bufB  = bufA + (size_t)N * GD_H;      // N*128 floats
    float* stats = bufB + (size_t)N * GD_H;      // 256 floats
    unsigned short* wp = (unsigned short*)(stats + 256);
    unsigned short* w1hi = wp;             wp += GD_IN * GD_H;
    unsigned short* w1lo = wp;             wp += GD_IN * GD_H;
    unsigned short* w2hi = wp;             wp += GD_IN * GD_H;
    unsigned short* w2lo = wp;             wp += GD_IN * GD_H;
    unsigned short* w3hi = wp;             wp += GD_IN * GD_OUT;
    unsigned short* w3lo = wp;
    float* out = (float*)d_out;

    // bf16 hi/lo activation planes alias the fp32 buffers (same footprint)
    unsigned short* hiA = (unsigned short*)bufA;
    unsigned short* loA = hiA + (size_t)N * GD_H;
    unsigned short* hiB = (unsigned short*)bufB;
    unsigned short* loB = hiB + (size_t)N * GD_H;

    const float invN = 1.0f / (float)N;
    const size_t NH = (size_t)N * GD_H;
    auto cdiv = [](long long a, long long b) { return (int)((a + b - 1) / b); };

    const int waves = cdiv(N, 16);
    const int gemmBlocks = cdiv((long long)waves * 32, 256);
    const int scatterBlocks = cdiv((long long)E * 32, 256);
    const int elemBlocksH = cdiv((long long)NH, 256);
    const int elemBlocksO = cdiv((long long)N * GD_OUT, 256);

    // ---- degrees / normalization ----
    fill_kernel<<<256, 256, 0, stream>>>(dinv, 1.0f, (size_t)N);  // self-loop weight
    deg_kernel<<<cdiv(E, 256), 256, 0, stream>>>(dst, ew, dinv, E);
    rsqrt_kernel<<<cdiv(N, 256), 256, 0, stream>>>(dinv, N);

    // ---- pack weights once (fragment-ordered bf16 hi/lo planes) ----
    pack_w<GD_H  ><<<cdiv(GD_IN * GD_H,   256), 256, 0, stream>>>(W1, w1hi, w1lo);
    pack_w<GD_H  ><<<cdiv(GD_IN * GD_H,   256), 256, 0, stream>>>(W2, w2hi, w2lo);
    pack_w<GD_OUT><<<cdiv(GD_IN * GD_OUT, 256), 256, 0, stream>>>(W3, w3hi, w3lo);

    // ---- layer 1: planes(A) -> conv(B) -> agg(A) -> planes(B) ----
    split_hilo<<<elemBlocksH, 256, 0, stream>>>(x, hiA, loA, NH);
    gemm_bf16x3<GD_H><<<gemmBlocks, 256, 0, stream>>>(hiA, loA, w1hi, w1lo, bufB, N);
    init_agg<GD_H><<<elemBlocksH, 256, 0, stream>>>(bufA, bufB, dinv, b1, N);
    scatter_edges<GD_H><<<scatterBlocks, 256, 0, stream>>>(src, dst, ew, dinv, bufB, bufA, E);
    fill_kernel<<<1, 256, 0, stream>>>(stats, 0.0f, 256);
    bn_stats<<<cdiv(N, 256), dim3(128, 2), 0, stream>>>(bufA, stats, N);
    bn_relu_split<<<elemBlocksH, 256, 0, stream>>>(bufA, hiB, loB, stats, g1, be1, N, invN);

    // ---- layer 2: planes(B) -> conv(A) -> agg(B) -> planes(A) ----
    gemm_bf16x3<GD_H><<<gemmBlocks, 256, 0, stream>>>(hiB, loB, w2hi, w2lo, bufA, N);
    init_agg<GD_H><<<elemBlocksH, 256, 0, stream>>>(bufB, bufA, dinv, b2, N);
    scatter_edges<GD_H><<<scatterBlocks, 256, 0, stream>>>(src, dst, ew, dinv, bufA, bufB, E);
    fill_kernel<<<1, 256, 0, stream>>>(stats, 0.0f, 256);
    bn_stats<<<cdiv(N, 256), dim3(128, 2), 0, stream>>>(bufB, stats, N);
    bn_relu_split<<<elemBlocksH, 256, 0, stream>>>(bufB, hiA, loA, stats, g2, be2, N, invN);

    // ---- layer 3: planes(A) -> conv(B, N x 64) -> d_out ----
    gemm_bf16x3<GD_OUT><<<gemmBlocks, 256, 0, stream>>>(hiA, loA, w3hi, w3lo, bufB, N);
    init_agg<GD_OUT><<<elemBlocksO, 256, 0, stream>>>(out, bufB, dinv, b3, N);
    scatter_edges<GD_OUT><<<scatterBlocks, 256, 0, stream>>>(src, dst, ew, dinv, bufB, out, E);
}
